// nchwAttentionLePE_31817117728874
// MI455X (gfx1250) — matrix operand order repaired and checked by reading, not compiled, not run on hardware
//
#include <hip/hip_runtime.h>

// ---------------------------------------------------------------------------
// CDNA5 (gfx1250, wave32) attention + LePE block.
// Strategy: one-time bf16 conversion into WMMA-operand-matched layouts, then
// all contractions are pure global_load_b128 -> v_wmma_f32_16x16x32_bf16.
// Attention stages K/V tiles into LDS with global_load_async_to_lds_b128.
// ---------------------------------------------------------------------------

#define BATCH    8
#define CDIM     256
#define NSP      1024      // H*W = 32*32
#define NHEADS   8
#define HDIM     32
#define ATT_SCALE 0.17677669529663687f   // 32^-0.5
#define LOG2E     1.4426950408889634f

#define USE_ASYNC_LDS 1    // global_load_async_to_lds_b128 via inline asm

typedef __bf16 bf16_t;
typedef bf16_t v16bf __attribute__((ext_vector_type(16)));
typedef float  v8f   __attribute__((ext_vector_type(8)));

union FragU { v16bf v; unsigned u[8]; uint4 q[2]; };
union AccU  { v8f   v; float    f[8]; };

__device__ __forceinline__ unsigned short f2bf(float f) {
  unsigned u = __float_as_uint(f);
  u += 0x7FFFu + ((u >> 16) & 1u);          // round-to-nearest-even
  return (unsigned short)(u >> 16);
}
__device__ __forceinline__ float bf2f(unsigned short h) {
  return __uint_as_float((unsigned)h << 16);
}
__device__ __forceinline__ unsigned pk2(unsigned short lo, unsigned short hi) {
  return (unsigned)lo | ((unsigned)hi << 16);
}
// packed f32x2 -> bf16x2 (v_cvt_pk_bf16_f32 when available)
__device__ __forceinline__ unsigned pkcvt(float a, float b) {
#if __has_builtin(__builtin_amdgcn_cvt_pk_bf16_f32)
  typedef __bf16 v2bf __attribute__((ext_vector_type(2)));
  union { v2bf v; unsigned u; } c;
  c.v = __builtin_amdgcn_cvt_pk_bf16_f32(a, b);
  return c.u;
#else
  return pk2(f2bf(a), f2bf(b));
#endif
}
// 16-bit A-matrix 16x32 layout (ISA 7.12.2): vgpr j holds K-pair
__device__ __forceinline__ int akk(int j, int half) {
  return ((j < 4) ? (2 * j) : (16 + 2 * (j - 4))) + 8 * half;
}
// 16-bit B-matrix 32x16 layout: same K order as A (vgpr j holds K-pair akk(j,half))
__device__ __forceinline__ int bkk(int j, int half) { return akk(j, half); }

__device__ __forceinline__ v8f wmma_bf16(v16bf a, v16bf b, v8f c) {
  return __builtin_amdgcn_wmma_f32_16x16x32_bf16(false, a, false, b,
                                                 (short)0, c, false, false);
}

// async 16B global->LDS copy (ASYNCcnt); fallback: synchronous copy
__device__ __forceinline__ void cp16_g2l(unsigned lds_off, const void* g) {
#if USE_ASYNC_LDS
  asm volatile("global_load_async_to_lds_b128 %0, %1, off"
               :: "v"(lds_off), "v"((unsigned long long)g) : "memory");
#else
  (void)lds_off; (void)g;
#endif
}
__device__ __forceinline__ void cp_wait_async() {
#if USE_ASYNC_LDS
  asm volatile("s_wait_asynccnt 0" ::: "memory");
#endif
}

// ---------------------------------------------------------------------------
// bulk fp32 -> bf16 conversion (same layout), float4-vectorized
// ---------------------------------------------------------------------------
__global__ __launch_bounds__(256)
void cvt_kernel(const float* __restrict__ in, unsigned short* __restrict__ out,
                int n4) {
  int i = blockIdx.x * 256 + threadIdx.x;
  if (i < n4) {
    float4 f = ((const float4*)in)[i];
    ((unsigned*)out)[2 * i]     = pkcvt(f.x, f.y);
    ((unsigned*)out)[2 * i + 1] = pkcvt(f.z, f.w);
  }
}

// ---------------------------------------------------------------------------
// x [b][c][n] fp32  ->  xt [b][n][c] bf16   (32x32 LDS tile transpose)
// ---------------------------------------------------------------------------
__global__ __launch_bounds__(256)
void xpose_kernel(const float* __restrict__ x, unsigned short* __restrict__ xt) {
  const int b = blockIdx.z, c0 = blockIdx.y * 32, n0 = blockIdx.x * 32;
  __shared__ unsigned short t[32][33];
  const int tc = threadIdx.x & 31, tr = threadIdx.x >> 5;   // tr in 0..7
  const float* xb = x + ((size_t)b * CDIM + c0) * NSP + n0;
#pragma unroll
  for (int i = 0; i < 32; i += 8)
    t[tr + i][tc] = f2bf(xb[(size_t)(tr + i) * NSP + tc]);  // t[c][n]
  __syncthreads();
  unsigned short* o = xt + ((size_t)b * NSP + n0) * CDIM + c0;
#pragma unroll
  for (int i = 0; i < 32; i += 8)
    o[(size_t)(tr + i) * CDIM + tc] = t[tc][tr + i];        // out[n][c]
}

// ---------------------------------------------------------------------------
// LDS-free WMMA GEMM: Out[b][m][n] = sum_k A[m][k] * Bt[b][n][k]
// A [M][K] bf16 row-major -> A-fragment = 2x global_load_b128 per tile
// Bt [N][K] bf16 row-major -> B-fragment = 2x global_load_b128 per tile
// mode 0: Out fp32 + bias.  mode 1: scatter bf16 into qT/kT/vN layouts.
// ---------------------------------------------------------------------------
__global__ __launch_bounds__(256)
void wmma_gemm_kernel(const unsigned short* __restrict__ A,
                      const unsigned short* __restrict__ Bt_all,
                      float* __restrict__ Outb,
                      const float* __restrict__ bias,
                      unsigned short* __restrict__ qT,   // [b*8+h][n][d]
                      unsigned short* __restrict__ kT,   // [b*8+h][m][d]
                      unsigned short* __restrict__ vN,   // [b*8+h][d][m]
                      int M, int K, int N, int mode)
{
  const int b     = blockIdx.z;
  const int m_blk = blockIdx.y * 128;
  const int n_blk = blockIdx.x * 128;
  const unsigned short* __restrict__ Bt = Bt_all + (size_t)b * N * K;

  const int tid  = threadIdx.x;
  const int wave = tid >> 5, lane = tid & 31;
  const int half = lane >> 4, l15 = lane & 15;
  const int wrow = (wave >> 2) * 64;
  const int wcol = (wave & 3) * 32;

  AccU acc[4][2];
#pragma unroll
  for (int tm = 0; tm < 4; ++tm)
#pragma unroll
    for (int tn = 0; tn < 2; ++tn)
#pragma unroll
      for (int i = 0; i < 8; ++i) acc[tm][tn].f[i] = 0.0f;

  for (int k0 = 0; k0 < K; k0 += 32) {
    FragU bfr[2];
#pragma unroll
    for (int tn = 0; tn < 2; ++tn) {
      const unsigned short* p =
          Bt + (size_t)(n_blk + wcol + tn * 16 + l15) * K + k0 + 8 * half;
      bfr[tn].q[0] = *(const uint4*)p;
      bfr[tn].q[1] = *(const uint4*)(p + 16);
    }
#pragma unroll
    for (int tm = 0; tm < 4; ++tm) {
      FragU af;
      const unsigned short* p =
          A + (size_t)(m_blk + wrow + tm * 16 + l15) * K + k0 + 8 * half;
      af.q[0] = *(const uint4*)p;
      af.q[1] = *(const uint4*)(p + 16);
#pragma unroll
      for (int tn = 0; tn < 2; ++tn)
        acc[tm][tn].v = wmma_bf16(af.v, bfr[tn].v, acc[tm][tn].v);
    }
  }

  if (mode == 0) {
#pragma unroll
    for (int tm = 0; tm < 4; ++tm)
#pragma unroll
      for (int tn = 0; tn < 2; ++tn)
#pragma unroll
        for (int i = 0; i < 8; ++i) {
          int gm = m_blk + wrow + tm * 16 + i + 8 * half;
          int gn = n_blk + wcol + tn * 16 + l15;
          Outb[(size_t)b * M * N + (size_t)gm * N + gn] =
              acc[tm][tn].f[i] + bias[gm];
        }
  } else {
    // scatter q/k/v into attention-friendly bf16 layouts
#pragma unroll
    for (int tm = 0; tm < 4; ++tm)
#pragma unroll
      for (int tn = 0; tn < 2; ++tn)
#pragma unroll
        for (int i = 0; i < 8; ++i) {
          int o  = m_blk + wrow + tm * 16 + i + 8 * half;
          int gn = n_blk + wcol + tn * 16 + l15;
          unsigned short hv = f2bf(acc[tm][tn].f[i]);
          if (o < CDIM) {                       // Q -> qT[bh][n][d]
            int h = o >> 5, d = o & 31;
            qT[(((size_t)b * NHEADS + h) * NSP + gn) * HDIM + d] = hv;
          } else if (o < 2 * CDIM) {            // K -> kT[bh][m][d]
            int oo = o - CDIM;
            int h = oo >> 5, d = oo & 31;
            kT[(((size_t)b * NHEADS + h) * NSP + gn) * HDIM + d] = hv;
          } else {                              // V -> vN[bh][d][m]
            int oo = o - 2 * CDIM;
            int h = oo >> 5, d = oo & 31;
            vN[(((size_t)b * NHEADS + h) * HDIM + d) * NSP + gn] = hv;
          }
        }
  }
}

// ---------------------------------------------------------------------------
// Flash attention. Block: one (b,h) x 128 query rows (8 waves x 16 rows).
// K/V tiles async-copied to LDS; S tile = one WMMA (K = HDIM = 32).
// Writes yT[b][n][c] bf16 (proj-GEMM B-operand layout).
// ---------------------------------------------------------------------------
__global__ __launch_bounds__(256)
void attn_kernel(const unsigned short* __restrict__ qT,
                 const unsigned short* __restrict__ kT,
                 const unsigned short* __restrict__ vN,
                 unsigned short* __restrict__ yT)
{
  const int b  = blockIdx.z;
  const int h  = blockIdx.y;
  const int qc = blockIdx.x;
  const int tid  = threadIdx.x;
  const int wave = tid >> 5, lane = tid & 31;
  const int half = lane >> 4, l15 = lane & 15;
  const int n0   = qc * 128 + wave * 16;
  const size_t bh = (size_t)b * NHEADS + h;

  const unsigned short* __restrict__ qTb = qT + bh * NSP * HDIM;
  const unsigned short* __restrict__ kTb = kT + bh * NSP * HDIM;
  const unsigned short* __restrict__ vNb = vN + bh * HDIM * NSP;

  __shared__ __attribute__((aligned(16))) unsigned short Ks[256][HDIM]; // [m][d]
  __shared__ __attribute__((aligned(16))) unsigned short Vs[HDIM][256]; // [d][m]
  __shared__ __attribute__((aligned(16))) unsigned short Pb[8][16][32]; // per-wave P

  // Persistent Q^T A-fragment straight from global (row-major [n][d])
  FragU aQ;
  {
    const unsigned short* p = qTb + (size_t)(n0 + l15) * HDIM + 8 * half;
    aQ.q[0] = *(const uint4*)p;
    aQ.q[1] = *(const uint4*)(p + 16);
  }

  AccU o0, o1;
  float rmax[8], rsum[8];
#pragma unroll
  for (int i = 0; i < 8; ++i) {
    o0.f[i] = 0.0f; o1.f[i] = 0.0f;
    rmax[i] = -1e30f; rsum[i] = 0.0f;
  }

  const unsigned ks_off = (unsigned)(unsigned long long)&Ks[0][0];
  const unsigned vs_off = (unsigned)(unsigned long long)&Vs[0][0];

  for (int stage = 0; stage < 4; ++stage) {
    const int ms = stage * 256;
    // ---- K stage: one contiguous 16KB chunk of kT ----
    {
      const char* kg = (const char*)(kTb + (size_t)ms * HDIM);
#pragma unroll
      for (int it = 0; it < 4; ++it) {
        unsigned off = (unsigned)(tid + it * 256) * 16u;   // 1024 x 16B
#if USE_ASYNC_LDS
        cp16_g2l(ks_off + off, kg + off);
#else
        *(uint4*)((char*)&Ks[0][0] + off) = *(const uint4*)(kg + off);
#endif
      }
    }
    // ---- V stage: 32 rows of 512B (global row stride 2KB) ----
    {
#pragma unroll
      for (int it = 0; it < 4; ++it) {
        int idx = tid + it * 256;                 // 1024 x 16B
        int d = idx >> 5;
        unsigned off = (unsigned)(idx & 31) * 16u;
        const char* vg = (const char*)(vNb + (size_t)d * NSP + ms) + off;
#if USE_ASYNC_LDS
        cp16_g2l(vs_off + (unsigned)d * 512u + off, vg);
#else
        *(uint4*)((char*)&Vs[d][0] + off) = *(const uint4*)vg;
#endif
      }
    }
    cp_wait_async();
    __syncthreads();

    for (int mb = 0; mb < 8; ++mb) {
      const int lm = mb * 32;
      FragU bK0, bK1;
#pragma unroll
      for (int j = 0; j < 8; ++j) {
        int kd = bkk(j, half);
        bK0.u[j] = *(const unsigned*)&Ks[lm + l15][kd];
        bK1.u[j] = *(const unsigned*)&Ks[lm + 16 + l15][kd];
      }
      v8f zacc = {};
      AccU s0, s1;
      s0.v = wmma_bf16(aQ.v, bK0.v, zacc);
      s1.v = wmma_bf16(aQ.v, bK1.v, zacc);

      // online softmax over this 32-key block
#pragma unroll
      for (int i = 0; i < 8; ++i) {
        float bm = fmaxf(s0.f[i], s1.f[i]);
#pragma unroll
        for (int off = 1; off < 16; off <<= 1)
          bm = fmaxf(bm, __shfl_xor(bm, off, 32));
        float mnew  = fmaxf(rmax[i], bm * ATT_SCALE);
        float alpha = exp2f((rmax[i] - mnew) * LOG2E);
        float e0 = exp2f((s0.f[i] * ATT_SCALE - mnew) * LOG2E);
        float e1 = exp2f((s1.f[i] * ATT_SCALE - mnew) * LOG2E);
        float ps = e0 + e1;
#pragma unroll
        for (int off = 1; off < 16; off <<= 1)
          ps += __shfl_xor(ps, off, 32);
        rsum[i] = rsum[i] * alpha + ps;
        rmax[i] = mnew;
        o0.f[i] *= alpha;
        o1.f[i] *= alpha;
        Pb[wave][i + 8 * half][l15]      = f2bf(e0);
        Pb[wave][i + 8 * half][16 + l15] = f2bf(e1);
      }
      __builtin_amdgcn_wave_barrier();
      asm volatile("s_wait_dscnt 0" ::: "memory");  // per-wave DS RAW fence

      FragU aP, bV0, bV1;
#pragma unroll
      for (int j = 0; j < 8; ++j)
        aP.u[j] = *(const unsigned*)&Pb[wave][l15][akk(j, half)];
#pragma unroll
      for (int j = 0; j < 8; ++j) {
        int km = bkk(j, half);
        bV0.u[j] = *(const unsigned*)&Vs[l15][lm + km];
        bV1.u[j] = *(const unsigned*)&Vs[16 + l15][lm + km];
      }
      o0.v = wmma_bf16(aP.v, bV0.v, o0.v);
      o1.v = wmma_bf16(aP.v, bV1.v, o1.v);
    }
    __syncthreads();
  }

  // epilogue -> yT[b][n][c] bf16, c = h*32 + d
#pragma unroll
  for (int i = 0; i < 8; ++i) {
    float inv = 1.0f / rsum[i];
    int n = n0 + i + 8 * half;
    unsigned short* yrow = yT + ((size_t)b * NSP + n) * CDIM + h * HDIM;
    yrow[l15]      = f2bf(o0.f[i] * inv);
    yrow[16 + l15] = f2bf(o1.f[i] * inv);
  }
}

// ---------------------------------------------------------------------------
// LePE depthwise 5x5 on v (bf16) + bias, accumulated into yT[b][n][c].
// ---------------------------------------------------------------------------
__global__ __launch_bounds__(256)
void lepe_kernel(const unsigned short* __restrict__ vN,   // flat [b][c][n] bf16
                 const float* __restrict__ w,             // [C][1][5][5]
                 const float* __restrict__ bias,          // [C]
                 unsigned short* __restrict__ yT)         // [b][n][c] bf16
{
  int idx = blockIdx.x * 256 + threadIdx.x;
  int px = idx & 31;
  int t  = idx >> 5;
  int py = t & 31;  t >>= 5;
  int c  = t & 255;
  int b  = t >> 8;
  const unsigned short* __restrict__ v = vN + ((size_t)b * CDIM + c) * NSP;
  const float* __restrict__ wc = w + (size_t)c * 25;
  float s = bias[c];
#pragma unroll
  for (int ky = 0; ky < 5; ++ky) {
    int iy = py + ky - 2;
    if (iy < 0 || iy >= 32) continue;
#pragma unroll
    for (int kx = 0; kx < 5; ++kx) {
      int ix = px + kx - 2;
      if (ix < 0 || ix >= 32) continue;
      s += bf2f(v[iy * 32 + ix]) * wc[ky * 5 + kx];
    }
  }
  unsigned short* yp = yT + ((size_t)b * NSP + py * 32 + px) * CDIM + c;
  *yp = f2bf(bf2f(*yp) + s);
}

// ---------------------------------------------------------------------------
// Host launcher
// ---------------------------------------------------------------------------
extern "C" void kernel_launch(void* const* d_in, const int* in_sizes, int n_in,
                              void* d_out, int out_size, void* d_ws, size_t ws_size,
                              hipStream_t stream) {
  const float* x      = (const float*)d_in[0];
  const float* qkv_w  = (const float*)d_in[1];
  const float* proj_w = (const float*)d_in[2];
  const float* proj_b = (const float*)d_in[3];
  const float* lepe_w = (const float*)d_in[4];
  const float* lepe_b = (const float*)d_in[5];
  float* out = (float*)d_out;

  // bf16 workspace carve-up (ushort units)
  unsigned short* p    = (unsigned short*)d_ws;
  unsigned short* wqkv = p;  p += (size_t)3 * CDIM * CDIM;         // [768][256]
  unsigned short* wprj = p;  p += (size_t)CDIM * CDIM;             // [256][256]
  unsigned short* xt   = p;  p += (size_t)BATCH * NSP * CDIM;      // [b][n][c]
  unsigned short* qT   = p;  p += (size_t)BATCH * CDIM * NSP;      // [bh][n][d]
  unsigned short* kT   = p;  p += (size_t)BATCH * CDIM * NSP;      // [bh][m][d]
  unsigned short* vN   = p;  p += (size_t)BATCH * CDIM * NSP;      // [bh][d][m]
  unsigned short* yT   = p;  p += (size_t)BATCH * NSP * CDIM;      // [b][n][c]

  dim3 blk(256);
  // one-time bf16 conversions
  cvt_kernel<<<dim3((3 * CDIM * CDIM / 4 + 255) / 256), blk, 0, stream>>>(
      qkv_w, wqkv, 3 * CDIM * CDIM / 4);
  cvt_kernel<<<dim3((CDIM * CDIM / 4 + 255) / 256), blk, 0, stream>>>(
      proj_w, wprj, CDIM * CDIM / 4);
  xpose_kernel<<<dim3(NSP / 32, CDIM / 32, BATCH), blk, 0, stream>>>(x, xt);

  // qkv GEMM (scatter epilogue into qT/kT/vN)
  wmma_gemm_kernel<<<dim3(NSP / 128, 768 / 128, BATCH), blk, 0, stream>>>(
      wqkv, xt, nullptr, nullptr, qT, kT, vN, 3 * CDIM, CDIM, NSP, 1);
  // attention -> yT
  attn_kernel<<<dim3(NSP / 128, NHEADS, BATCH), blk, 0, stream>>>(qT, kT, vN, yT);
  // yT += lepe(v)
  lepe_kernel<<<dim3((BATCH * CDIM * NSP) / 256), blk, 0, stream>>>(
      vN, lepe_w, lepe_b, yT);
  // out = W_proj @ y + b (fp32 epilogue)
  wmma_gemm_kernel<<<dim3(NSP / 128, CDIM / 128, BATCH), blk, 0, stream>>>(
      wprj, yT, out, proj_b, nullptr, nullptr, nullptr, CDIM, CDIM, NSP, 0);
}
